// GCNIIBackbone_55817394979591
// MI455X (gfx1250) — matrix operand order repaired and checked by reading, not compiled, not run on hardware
//
#include <hip/hip_runtime.h>
#include <math.h>

typedef __attribute__((ext_vector_type(16))) __bf16 v16bf;
typedef __attribute__((ext_vector_type(8)))  float  v8f;

#define DF 128
#define ALPHA 0.5f
#define BN_EPS 1e-5f

// ---------------- graph preprocessing ----------------
__global__ void deg_kernel(const int* __restrict__ dst, float* __restrict__ deg, int Et) {
    int e = blockIdx.x * 256 + threadIdx.x;
    if (e < Et) atomicAdd(&deg[dst[e]], 1.0f);
}

__global__ void dis_kernel(float* __restrict__ deg, int N) {
    int i = blockIdx.x * 256 + threadIdx.x;
    if (i < N) {
        float d = deg[i];
        deg[i] = (d > 0.0f) ? rsqrtf(fmaxf(d, 1.0f)) : 0.0f;
    }
}

__global__ void norm_kernel(const int* __restrict__ src, const int* __restrict__ dst,
                            const float* __restrict__ dis, float* __restrict__ norm, int Et) {
    int e = blockIdx.x * 256 + threadIdx.x;
    if (e < Et) norm[e] = dis[src[e]] * dis[dst[e]] * (1.0f - ALPHA);  // fold (1-alpha)
}

// ---------------- weight prep: transpose + bf16 ----------------
// Wtp[n*128+k]    = w_proj[k,n]
// Wtl[l][n*256+k] = k<128 ? w1[l,k,n] : w2[l,k-128,n]
__global__ void prep_w(const float* __restrict__ wp, const float* __restrict__ w1,
                       const float* __restrict__ w2, __bf16* __restrict__ Wtp,
                       __bf16* __restrict__ Wtl) {
    int t = blockIdx.x * 256 + threadIdx.x;
    if (t < DF * DF) {
        int n = t >> 7, k = t & 127;
        Wtp[n * DF + k] = (__bf16)wp[k * DF + n];
        return;
    }
    int t2 = t - DF * DF;
    if (t2 < 4 * DF * 256) {
        int l = t2 / (DF * 256);
        int r = t2 - l * (DF * 256);
        int n = r >> 8, k = r & 255;
        float v = (k < DF) ? w1[(l * DF + k) * DF + n] : w2[(l * DF + (k - DF)) * DF + n];
        Wtl[(size_t)l * DF * 256 + n * 256 + k] = (__bf16)v;
    }
}

// ---------------- WMMA helpers ----------------
// A fragment: 16x32 bf16 from an fp32 row. Lane holds 16 contiguous K at kb,
// with kb = kt*32 + (lane>=16 ? 16 : 0).
__device__ inline v16bf load_a_frag(const float* p) {
    v8f a0 = *(const v8f*)(p);
    v8f a1 = *(const v8f*)(p + 8);
    v16bf f;
#pragma unroll
    for (int i = 0; i < 8; i++) { f[i] = (__bf16)a0[i]; f[8 + i] = (__bf16)a1[i]; }
    return f;
}

// ---------------- projection GEMM: h = x@Wp + b ; h0s = alpha*h ----------------
__global__ __launch_bounds__(256) void gemm_proj(const float* __restrict__ A,
                                                 const __bf16* __restrict__ Wt,
                                                 const float* __restrict__ bias,
                                                 float* __restrict__ h,
                                                 float* __restrict__ h0s, int N) {
    __shared__ __bf16 Bs[DF * DF];  // 32 KB, shared by all 8 waves
    {
        const uint4* sv = (const uint4*)Wt;
        uint4* dv = (uint4*)Bs;
#pragma unroll
        for (int i = threadIdx.x; i < DF * DF * 2 / 16; i += 256) dv[i] = sv[i];
    }
    __syncthreads();

    int lane = threadIdx.x & 31;
    int wave = threadIdx.x >> 5;
    int rbase = blockIdx.x * 128 + wave * 16;
    int mrow = rbase + (lane & 15);
    if (mrow > N - 1) mrow = N - 1;
    bool hi = lane >= 16;
    const float* arow = A + (size_t)mrow * DF;

    v8f cacc[8] = {};
#pragma unroll
    for (int kt = 0; kt < 4; kt++) {
        v16bf af = load_a_frag(arow + kt * 32 + (hi ? 16 : 0));
        int kbB = kt * 32 + (hi ? 16 : 0);
#pragma unroll
        for (int nt = 0; nt < 8; nt++) {
            int n = nt * 16 + (lane & 15);
            v16bf bf = *(const v16bf*)(Bs + n * DF + kbB);
            cacc[nt] = __builtin_amdgcn_wmma_f32_16x16x32_bf16(
                false, af, false, bf, (short)0, cacc[nt], false, false);
        }
    }
    int col0 = lane & 15;
    int radd = hi ? 8 : 0;
#pragma unroll
    for (int nt = 0; nt < 8; nt++) {
        int col = nt * 16 + col0;
        float bv = bias[col];
#pragma unroll
        for (int v = 0; v < 8; v++) {
            int r = rbase + radd + v;
            if (r < N) {
                float val = cacc[nt][v] + bv;
                h[(size_t)r * DF + col] = val;
                h0s[(size_t)r * DF + col] = ALPHA * val;
            }
        }
    }
}

// ---------------- layer GEMM (K=256 fused): out = (1-b)*(Xs+X0s) + b*(Xs@w1 + X0s@w2)
// out aliases Xs (safe: all A reads of a wave's rows precede its epilogue stores).
__global__ __launch_bounds__(256) void gemm_layer(const float* Xs,
                                                  const float* __restrict__ X0s,
                                                  const __bf16* __restrict__ Wt,
                                                  float* out, int N, float beta) {
    __shared__ __bf16 Bs[DF * 256];  // 64 KB, shared by all 8 waves
    {
        const uint4* sv = (const uint4*)Wt;
        uint4* dv = (uint4*)Bs;
#pragma unroll
        for (int i = threadIdx.x; i < DF * 256 * 2 / 16; i += 256) dv[i] = sv[i];
    }
    __syncthreads();

    int lane = threadIdx.x & 31;
    int wave = threadIdx.x >> 5;
    int rbase = blockIdx.x * 128 + wave * 16;
    int mrow = rbase + (lane & 15);
    if (mrow > N - 1) mrow = N - 1;
    bool hi = lane >= 16;
    const float* ar1 = Xs + (size_t)mrow * DF;
    const float* ar2 = X0s + (size_t)mrow * DF;

    v8f cacc[8] = {};
#pragma unroll
    for (int kt = 0; kt < 8; kt++) {
        const float* arow = (kt < 4) ? ar1 : ar2;
        int kof = (kt & 3) * 32;
        v16bf af = load_a_frag(arow + kof + (hi ? 16 : 0));
        int kbB = kt * 32 + (hi ? 16 : 0);
#pragma unroll
        for (int nt = 0; nt < 8; nt++) {
            int n = nt * 16 + (lane & 15);
            v16bf bf = *(const v16bf*)(Bs + n * 256 + kbB);
            cacc[nt] = __builtin_amdgcn_wmma_f32_16x16x32_bf16(
                false, af, false, bf, (short)0, cacc[nt], false, false);
        }
    }
    float omb = 1.0f - beta;
    int col0 = lane & 15;
    int radd = hi ? 8 : 0;
#pragma unroll
    for (int nt = 0; nt < 8; nt++) {
        int col = nt * 16 + col0;
#pragma unroll
        for (int v = 0; v < 8; v++) {
            int r = rbase + radd + v;
            if (r < N) {
                size_t idx = (size_t)r * DF + col;
                float lin = Xs[idx] + X0s[idx];
                out[idx] = omb * lin + beta * cacc[nt][v];
            }
        }
    }
}

// ---------------- SpMM scatter: agg[dst] += h[src] * norm ----------------
__global__ void spmm_scatter(const int* __restrict__ src, const int* __restrict__ dst,
                             const float* __restrict__ norm, const float* __restrict__ h,
                             float* __restrict__ agg, int Et) {
    long long t = (long long)blockIdx.x * 256 + threadIdx.x;
    int e = (int)(t >> 5);
    if (e >= Et) return;
    int c = (int)(t & 31) * 4;
    float w = norm[e];
    int s = src[e], d = dst[e];
    float4 v = *(const float4*)(h + (size_t)s * DF + c);
    float* ap = agg + (size_t)d * DF + c;
    atomicAdd(ap + 0, v.x * w);
    atomicAdd(ap + 1, v.y * w);
    atomicAdd(ap + 2, v.z * w);
    atomicAdd(ap + 3, v.w * w);
}

// ---------------- BatchNorm: column stats then apply+ReLU ----------------
__global__ __launch_bounds__(128) void bn_stats(const float* __restrict__ hp,
                                                float* __restrict__ sums, int N) {
    int col = threadIdx.x;  // 128 columns
    float s = 0.0f, q = 0.0f;
    for (int i = blockIdx.x; i < N; i += gridDim.x) {
        float v = hp[(size_t)i * DF + col];
        s += v;
        q += v * v;
    }
    atomicAdd(&sums[col], s);
    atomicAdd(&sums[DF + col], q);
}

__global__ void bn_apply(const float* __restrict__ hp, const float* __restrict__ sums,
                         const float* __restrict__ gamma, const float* __restrict__ betab,
                         float* __restrict__ out, int N) {
    long long t = (long long)blockIdx.x * 256 + threadIdx.x;
    if (t >= (long long)N * DF) return;
    int j = (int)(t & (DF - 1));
    float invN = 1.0f / (float)N;
    float mu = sums[j] * invN;
    float var = sums[DF + j] * invN - mu * mu;
    float v = hp[t];
    float y = gamma[j] * (v - mu) * rsqrtf(var + BN_EPS) + betab[j];
    out[t] = fmaxf(y, 0.0f);
}

// ---------------- host orchestration ----------------
extern "C" void kernel_launch(void* const* d_in, const int* in_sizes, int n_in,
                              void* d_out, int out_size, void* d_ws, size_t ws_size,
                              hipStream_t stream) {
    const float* x      = (const float*)d_in[0];
    const int*   src    = (const int*)d_in[1];
    const int*   dst    = (const int*)d_in[2];
    const float* w_proj = (const float*)d_in[3];
    const float* b_proj = (const float*)d_in[4];
    const float* w1     = (const float*)d_in[5];
    const float* w2     = (const float*)d_in[6];
    const float* gamma  = (const float*)d_in[7];
    const float* betabn = (const float*)d_in[8];

    int N  = in_sizes[0] / DF;
    int Et = in_sizes[1];

    // carve workspace (512B aligned chunks)
    char* wsp = (char*)d_ws;
    auto carve = [&](size_t bytes) {
        void* p = (void*)wsp;
        wsp += (bytes + 511) & ~(size_t)511;
        return p;
    };
    float*  deg  = (float*)carve((size_t)N * 4);           // becomes dis in-place
    float*  norm = (float*)carve((size_t)Et * 4);
    float*  h    = (float*)carve((size_t)N * DF * 4);
    float*  h0s  = (float*)carve((size_t)N * DF * 4);
    float*  agg  = (float*)carve((size_t)N * DF * 4);      // reused as hpre in-place
    __bf16* Wtp  = (__bf16*)carve((size_t)DF * DF * 2);
    __bf16* Wtl  = (__bf16*)carve((size_t)4 * DF * 256 * 2);
    float*  bns  = (float*)carve(2 * DF * 4);

    int gE   = (Et + 255) / 256;
    int gN   = (N + 255) / 256;
    int gR   = (N + 127) / 128;                       // GEMM row blocks (128 rows/block)
    int gSp  = (int)(((long long)Et * 32 + 255) / 256);
    int gEl  = (int)(((long long)N * DF + 255) / 256);
    int gPw  = (DF * DF + 4 * DF * 256 + 255) / 256;

    // graph normalization
    hipMemsetAsync(deg, 0, (size_t)N * 4, stream);
    deg_kernel<<<gE, 256, 0, stream>>>(dst, deg, Et);
    dis_kernel<<<gN, 256, 0, stream>>>(deg, N);
    norm_kernel<<<gE, 256, 0, stream>>>(src, dst, deg, norm, Et);

    // weights -> bf16 transposed; projection GEMM
    prep_w<<<gPw, 256, 0, stream>>>(w_proj, w1, w2, Wtp, Wtl);
    gemm_proj<<<gR, 256, 0, stream>>>(x, Wtp, b_proj, h, h0s, N);

    for (int l = 0; l < 4; l++) {
        float beta = logf(1.0f / (float)(l + 1) + 1.0f);  // THETA = 1.0
        hipMemsetAsync(agg, 0, (size_t)N * DF * 4, stream);
        spmm_scatter<<<gSp, 256, 0, stream>>>(src, dst, norm, h, agg, Et);
        gemm_layer<<<gR, 256, 0, stream>>>(agg, h0s, Wtl + (size_t)l * DF * 256, agg, N, beta);
        hipMemsetAsync(bns, 0, 2 * DF * 4, stream);
        bn_stats<<<512, 128, 0, stream>>>(agg, bns, N);
        float* outp = (l == 3) ? (float*)d_out : h;
        bn_apply<<<gEl, 256, 0, stream>>>(agg, bns, gamma + l * DF, betabn + l * DF, outp, N);
    }
}
